// LocallyConnectedBlock_73607149519468
// MI455X (gfx1250) — matrix operand order repaired and checked by reading, not compiled, not run on hardware
//
#include <hip/hip_runtime.h>

// Grouped SIREN: 64 groups x (2 -> 256 -> 256 -> 256 -> 256 -> 3), sin(30x).
// Hidden layers via v_wmma_f32_16x16x32_f16 (f16 in, f32 accumulate).

#define G_GROUPS 64
#define N_PTS    4096
#define H_DIM    256
#define OUTF     3
#define OMEGA    30.0f
#define N_TILE   128          // rows per block (8 waves x 16 rows)
#define LDX      264          // padded activation row stride (halves): 528B/row, 16B aligned
#define LDW      264          // padded transposed-weight row stride (halves)

typedef __attribute__((ext_vector_type(16))) _Float16 v16h;
typedef __attribute__((ext_vector_type(8)))  _Float16 v8h;
typedef __attribute__((ext_vector_type(8)))  float    v8f;

__device__ __forceinline__ v16h cat8(v8h lo, v8h hi) {
  return __builtin_shufflevector(lo, hi, 0, 1, 2, 3, 4, 5, 6, 7,
                                         8, 9, 10, 11, 12, 13, 14, 15);
}

// Stage W[g] (H x H, row-major f32 in global) into LDS transposed as f16:
//   Wt[n*LDW + k] = (f16) W[k*H + n]
__device__ __forceinline__ void load_w_t(const float* __restrict__ Wg,
                                         _Float16* __restrict__ Wt, int tid) {
  for (int i = tid * 4; i < H_DIM * H_DIM; i += 256 * 4) {
    const float4 w = *(const float4*)(Wg + i);
    const int k = i >> 8;             // W row (K index)
    const int n = i & (H_DIM - 1);    // W col (N index), multiple of 4
    Wt[(n + 0) * LDW + k] = (_Float16)w.x;
    Wt[(n + 1) * LDW + k] = (_Float16)w.y;
    Wt[(n + 2) * LDW + k] = (_Float16)w.z;
    Wt[(n + 3) * LDW + k] = (_Float16)w.w;
  }
}

// One hidden layer on a 128x256 tile: Xout = sin(OMEGA*(Xin @ W + b)).
// Wave `wave` owns rows [wave*16, wave*16+16).
__device__ __forceinline__ void hidden_layer(const _Float16* __restrict__ Xin,
                                             _Float16* __restrict__ Xout,
                                             const _Float16* __restrict__ Wt,
                                             const float* __restrict__ bg,
                                             int wave, int lane) {
  const int half = lane >> 4;   // 0 or 1 (lane group)
  const int l15  = lane & 15;

  // A fragments (16x32 f16): lane group selects K offset 0/8; halves [0..7] hold
  // K = kb..kb+7, halves [8..15] hold K = kb+16..kb+23 (ISA 16-bit A layout).
  const int arow = wave * 16 + l15;
  const int akb  = half * 8;
  v16h afrag[8];
#pragma unroll
  for (int kt = 0; kt < 8; ++kt) {
    const _Float16* ap = Xin + arow * LDX + kt * 32 + akb;
    afrag[kt] = cat8(*(const v8h*)ap, *(const v8h*)(ap + 16));
  }

  // B fragments (32x16 f16): lane group selects K offset 0/8; halves [0..7] hold
  // K = kb..kb+7, halves [8..15] hold K = kb+16..kb+23 for col l15 (matches A).
  const int bkb = half * 8;
#pragma unroll 2
  for (int ct = 0; ct < 16; ++ct) {
    const int col = ct * 16 + l15;
    v8f acc = {};
#pragma unroll
    for (int kt = 0; kt < 8; ++kt) {
      const _Float16* bp = Wt + col * LDW + kt * 32 + bkb;
      const v16h bfrag = cat8(*(const v8h*)bp, *(const v8h*)(bp + 16));
      acc = __builtin_amdgcn_wmma_f32_16x16x32_f16(
          false, afrag[kt], false, bfrag, (short)0, acc, false, false);
    }
    // Epilogue per ISA C/D layout: VGPR m holds row (half*8 + m), col = l15.
    const float bias  = bg[col];
    const int   rbase = wave * 16 + half * 8;
#pragma unroll
    for (int m = 0; m < 8; ++m) {
      const float y = __sinf(OMEGA * (acc[m] + bias));
      Xout[(rbase + m) * LDX + col] = (_Float16)y;
    }
  }
}

__global__ __launch_bounds__(256, 1)
void siren_lc_kernel(const float* __restrict__ coords,
                     const float* __restrict__ W0, const float* __restrict__ b0,
                     const float* __restrict__ W1, const float* __restrict__ b1,
                     const float* __restrict__ W2, const float* __restrict__ b2,
                     const float* __restrict__ W3, const float* __restrict__ b3,
                     const float* __restrict__ W4, const float* __restrict__ b4,
                     float* __restrict__ out) {
  extern __shared__ _Float16 smem[];
  _Float16* Xa = smem;                  // 128 x LDX activations (ping)
  _Float16* Xb = Xa + N_TILE * LDX;     // 128 x LDX activations (pong)
  _Float16* Wt = Xb + N_TILE * LDX;     // 256 x LDW transposed weights

  const int tid  = threadIdx.x;
  const int wave = tid >> 5;
  const int lane = tid & 31;
  const int g    = blockIdx.x >> 5;             // 32 row-tiles per group
  const int r0   = (blockIdx.x & 31) * N_TILE;

  const float* cg  = coords + ((size_t)g * N_PTS + r0) * 2;
  const float* W0g = W0 + (size_t)g * 2 * H_DIM;
  const float* b0g = b0 + (size_t)g * H_DIM;
  const float* W1g = W1 + (size_t)g * H_DIM * H_DIM;
  const float* b1g = b1 + (size_t)g * H_DIM;
  const float* W2g = W2 + (size_t)g * H_DIM * H_DIM;
  const float* b2g = b2 + (size_t)g * H_DIM;
  const float* W3g = W3 + (size_t)g * H_DIM * H_DIM;
  const float* b3g = b3 + (size_t)g * H_DIM;
  const float* W4g = W4 + (size_t)g * H_DIM * OUTF;
  const float* b4g = b4 + (size_t)g * OUTF;

  // Warm L2 for the first hidden weight matrix (global_prefetch_b8).
  for (int off = tid * 64; off < H_DIM * H_DIM; off += 256 * 64)
    __builtin_prefetch(W1g + off, 0, 1);

  // ---- layer 0: [128,2] @ [2,256] + b0 -> sin (VALU; K too small for WMMA) ----
  for (int idx = tid; idx < N_TILE * H_DIM; idx += 256) {
    const int r = idx >> 8;
    const int h = idx & (H_DIM - 1);
    const float v = cg[r * 2 + 0] * W0g[h] + cg[r * 2 + 1] * W0g[H_DIM + h] + b0g[h];
    Xa[r * LDX + h] = (_Float16)__sinf(OMEGA * v);
  }
  load_w_t(W1g, Wt, tid);
  __syncthreads();

  // ---- hidden layer 1 ----
  for (int off = tid * 64; off < H_DIM * H_DIM; off += 256 * 64)
    __builtin_prefetch(W2g + off, 0, 1);
  hidden_layer(Xa, Xb, Wt, b1g, wave, lane);
  __syncthreads();

  // ---- hidden layer 2 ----
  load_w_t(W2g, Wt, tid);
  __syncthreads();
  for (int off = tid * 64; off < H_DIM * H_DIM; off += 256 * 64)
    __builtin_prefetch(W3g + off, 0, 1);
  hidden_layer(Xb, Xa, Wt, b2g, wave, lane);
  __syncthreads();

  // ---- hidden layer 3 ----
  load_w_t(W3g, Wt, tid);
  __syncthreads();
  hidden_layer(Xa, Xb, Wt, b3g, wave, lane);
  __syncthreads();

  // ---- layer 4: [128,256] @ [256,3] + b4, no activation (VALU) ----
  for (int idx = tid; idx < N_TILE * OUTF; idx += 256) {
    const int r = idx / OUTF;
    const int o = idx - r * OUTF;
    float s = b4g[o];
    const _Float16* xr = Xb + r * LDX;
#pragma unroll 8
    for (int k = 0; k < H_DIM; ++k)
      s += (float)xr[k] * W4g[k * OUTF + o];
    out[((size_t)g * N_PTS + r0 + r) * OUTF + o] = s;
  }
}

extern "C" void kernel_launch(void* const* d_in, const int* in_sizes, int n_in,
                              void* d_out, int out_size, void* d_ws, size_t ws_size,
                              hipStream_t stream) {
  (void)in_sizes; (void)n_in; (void)out_size; (void)d_ws; (void)ws_size;
  const float* coords = (const float*)d_in[0];
  const float* W0 = (const float*)d_in[1];
  const float* b0 = (const float*)d_in[2];
  const float* W1 = (const float*)d_in[3];
  const float* b1 = (const float*)d_in[4];
  const float* W2 = (const float*)d_in[5];
  const float* b2 = (const float*)d_in[6];
  const float* W3 = (const float*)d_in[7];
  const float* b3 = (const float*)d_in[8];
  const float* W4 = (const float*)d_in[9];
  const float* b4 = (const float*)d_in[10];
  float* out = (float*)d_out;

  const int smem_bytes = (2 * N_TILE * LDX + H_DIM * LDW) * (int)sizeof(_Float16); // 264 KB
  hipFuncSetAttribute(reinterpret_cast<const void*>(siren_lc_kernel),
                      hipFuncAttributeMaxDynamicSharedMemorySize, smem_bytes);

  dim3 grid(G_GROUPS * (N_PTS / N_TILE));  // 2048 blocks
  dim3 block(256);                         // 8 wave32s
  siren_lc_kernel<<<grid, block, smem_bytes, stream>>>(
      coords, W0, b0, W1, b1, W2, b2, W3, b3, W4, b4, out);
}
